// GNN_28441273434635
// MI455X (gfx1250) — matrix operand order repaired and checked by reading, not compiled, not run on hardware
//
#include <hip/hip_runtime.h>
#include <hip/hip_bf16.h>
#include <stdint.h>

#define IN_C  128
#define HID_C 256
#define OUT_C 128

typedef __attribute__((ext_vector_type(16))) __bf16         v16bf;
typedef __attribute__((ext_vector_type(16))) unsigned short v16us;
typedef __attribute__((ext_vector_type(8)))  float          v8f;
typedef __attribute__((ext_vector_type(4)))  unsigned int   v4u;
typedef __attribute__((ext_vector_type(8)))  int            v8i;
typedef __attribute__((ext_vector_type(4)))  int            v4i;

__device__ __forceinline__ unsigned short f32_to_bf16_bits(float f) {
    unsigned int u = __float_as_uint(f);
    u += 0x7FFFu + ((u >> 16) & 1u);          // round-to-nearest-even
    return (unsigned short)(u >> 16);
}

// ---------------- degree / normalization ----------------
__global__ void deg_init_kernel(float* __restrict__ deg, int n) {
    int i = blockIdx.x * blockDim.x + threadIdx.x;
    if (i < n) deg[i] = 1.0f;                 // self-loop contribution
}

__global__ void deg_count_kernel(const int* __restrict__ dst, float* __restrict__ deg, int e) {
    int i = blockIdx.x * blockDim.x + threadIdx.x;
    if (i < e) atomicAdd(&deg[dst[i]], 1.0f);
}

__global__ void dinv_kernel(const float* __restrict__ deg, float* __restrict__ dinv, int n) {
    int i = blockIdx.x * blockDim.x + threadIdx.x;
    if (i < n) dinv[i] = __frsqrt_rn(deg[i]); // deg >= 1 always
}

// ---------------- fp32 -> bf16 conversion (plain and transposing) ----------------
__global__ void to_bf16_kernel(const float* __restrict__ in, unsigned short* __restrict__ out, int n) {
    int i = blockIdx.x * blockDim.x + threadIdx.x;
    if (i < n) out[i] = f32_to_bf16_bits(in[i]);
}

// W[K][N] f32 (row-major) -> out[N][K] bf16 (i.e. B^T, K-contiguous)
template <int K, int N>
__global__ void wT_to_bf16_kernel(const float* __restrict__ W, unsigned short* __restrict__ out) {
    int idx = blockIdx.x * blockDim.x + threadIdx.x;
    if (idx >= K * N) return;
    int k = idx / N, n = idx % N;
    out[(size_t)n * K + k] = f32_to_bf16_bits(W[idx]);
}

// ---------------- TDM: 2D tile load -> LDS (D# per ISA 8.3/8.4) ----------------
#if __has_builtin(__builtin_amdgcn_tensor_load_to_lds)
#define USE_TDM 1
__device__ __forceinline__ void tdm_load_2d(unsigned lds_byte_off,
                                            const unsigned short* gptr,
                                            int tensor_d0, int tensor_d1,
                                            int tile_d0, int tile_d1,
                                            long long stride0_elems,
                                            int pad_interval_code, int pad_amount_code) {
    unsigned long long ga = (unsigned long long)(uintptr_t)gptr;
    v4u g0;
    g0[0] = 1u;                                            // count=1 (valid), user mode
    g0[1] = lds_byte_off;                                  // lds_addr
    g0[2] = (unsigned)(ga & 0xFFFFFFFFu);                  // global_addr[31:0]
    g0[3] = (unsigned)((ga >> 32) & 0x1FFFFFFu)            // global_addr[56:32]
          | (2u << 30);                                    // type = 2 ("image")
    v8i g1;
    g1[0] = (int)((1u << 16)                               // data_size = 1 -> 2 bytes
                | (1u << 20)                               // pad_enable
                | (((unsigned)pad_interval_code & 7u) << 22)
                | (((unsigned)pad_amount_code & 0x7Fu) << 25));
    g1[1] = (int)(((unsigned)tensor_d0 & 0xFFFFu) << 16);  // tensor_dim0[15:0]
    g1[2] = (int)((((unsigned)tensor_d0 >> 16) & 0xFFFFu)
                | (((unsigned)tensor_d1 & 0xFFFFu) << 16));
    g1[3] = (int)((((unsigned)tensor_d1 >> 16) & 0xFFFFu)
                | (((unsigned)tile_d0 & 0xFFFFu) << 16));  // tile_dim0
    g1[4] = (int)((unsigned)tile_d1 & 0xFFFFu);            // tile_dim1 (tile_dim2 = 0)
    g1[5] = (int)((unsigned long long)stride0_elems & 0xFFFFFFFFull);       // dim0 stride
    g1[6] = (int)(((unsigned long long)stride0_elems >> 32) & 0xFFFFull);
    g1[7] = 0;
    v4i z4 = {0, 0, 0, 0};
#if defined(__clang_major__) && __clang_major__ >= 23
    v8i z8 = {0, 0, 0, 0, 0, 0, 0, 0};
    __builtin_amdgcn_tensor_load_to_lds(g0, g1, z4, z4, z8, 0);
#else
    __builtin_amdgcn_tensor_load_to_lds(g0, g1, z4, z4, 0);
#endif
}
#else
#define USE_TDM 0
#endif

// ---------------- WMMA GEMM: C[MxN] = A[MxK](bf16) * BT[NxK](bf16), f32 accumulate
// grid = (N/64, M/16), block = 128 threads (4 wave32 waves); each wave -> one 16x16 tile.
// Whole K-strips of A (16xK) and B^T (64xK) staged once into LDS (TDM when available),
// padded row stride S = K+8 halfs: bank-staggered and 16B-aligned for ds_load_b128.
template <int K, int N>
__global__ __launch_bounds__(128)
void wmma_gemm_kernel(const unsigned short* __restrict__ A,
                      const unsigned short* __restrict__ BT,
                      float* __restrict__ C, int M) {
    constexpr int S = K + 8;                  // padded LDS row stride (halfs)
    __shared__ alignas(16) unsigned short At[16][S];
    __shared__ alignas(16) unsigned short Bt[64][S];

    const int tid  = threadIdx.x;
    const int wave = tid >> 5;
    const int lane = tid & 31;
    const int row0 = blockIdx.y * 16;
    const int col0 = blockIdx.x * 64;

#if USE_TDM
    // pad_interval: one row = K/2 DWORDs -> 2^(code+1) DWORDs; pad_amount: 4 DWORDs (code 3)
    constexpr int PAD_IVL = (K == 128) ? 5 : 6;
    if (tid < 32) {
        tdm_load_2d((unsigned)(uintptr_t)&At[0][0], A + (size_t)row0 * K,
                    K, M - row0, K, 16, (long long)K, PAD_IVL, 3);
        __builtin_amdgcn_s_wait_tensorcnt(0);
    } else if (tid < 64) {
        tdm_load_2d((unsigned)(uintptr_t)&Bt[0][0], BT + (size_t)col0 * K,
                    K, N - col0, K, 64, (long long)K, PAD_IVL, 3);
        __builtin_amdgcn_s_wait_tensorcnt(0);
    }
#else
    // cooperative fallback: 128-bit strip loads
    for (int u = tid; u < 16 * (K / 8); u += 128) {
        int r = u / (K / 8), g = u % (K / 8);
        *(uint4*)&At[r][g * 8] = *(const uint4*)(A + (size_t)(row0 + r) * K + g * 8);
    }
    for (int u = tid; u < 64 * (K / 8); u += 128) {
        int r = u / (K / 8), g = u % (K / 8);
        *(uint4*)&Bt[r][g * 8] = *(const uint4*)(BT + (size_t)(col0 + r) * K + g * 8);
    }
#endif
    __syncthreads();

    const int r    = lane & 15;               // A row / C col within tile
    const int half = lane >> 4;               // K/M half selector
    const int cN   = (lane & 15) + wave * 16; // column in the 64-wide block tile

    v8f acc = {};
    union Frag { uint4 q[2]; v16us v; };

#pragma unroll
    for (int kk = 0; kk < K; kk += 32) {
        // A fragment (ISA K-map i + 8*((i>>3)+half)): two contiguous 8-half runs
        Frag fa, fb;
        fa.q[0] = *(const uint4*)&At[r][kk + 8 * half];
        fa.q[1] = *(const uint4*)&At[r][kk + 16 + 8 * half];
        // B fragment (ISA K-map i + 8*((i>>3)+half)): two contiguous 8-half runs
        fb.q[0] = *(const uint4*)&Bt[cN][kk + 8 * half];
        fb.q[1] = *(const uint4*)&Bt[cN][kk + 16 + 8 * half];
        v16bf av = __builtin_bit_cast(v16bf, fa.v);
        v16bf bv = __builtin_bit_cast(v16bf, fb.v);
        acc = __builtin_amdgcn_wmma_f32_16x16x32_bf16(
                  false, av, false, bv, (short)0, acc, false, false);
    }

    // store C: VGPR v holds row (v + 8*half), col = lane&15
#pragma unroll
    for (int v = 0; v < 8; ++v) {
        const int rr = v + 8 * half;
        C[(size_t)(row0 + rr) * N + col0 + cN] = acc[v];
    }
}

// ---------------- self-loop init: AGG[i,c] = H[i,c] * dinv[i]^2 ----------------
template <int C>
__global__ void self_loop_init_kernel(const float* __restrict__ H, const float* __restrict__ dinv,
                                      float* __restrict__ AGG, int n_nodes) {
    long long idx = (long long)blockIdx.x * blockDim.x + threadIdx.x;
    long long total = (long long)n_nodes * C;
    if (idx >= total) return;
    int node = (int)(idx / C);
    float d = dinv[node];
    AGG[idx] = H[idx] * d * d;
}

// ---------------- edge scatter: AGG[dst] += H[src] * dinv[src]*dinv[dst] ----------------
template <int C>
__global__ void agg_edges_kernel(const float* __restrict__ H,
                                 const int* __restrict__ src, const int* __restrict__ dst,
                                 const float* __restrict__ dinv,
                                 float* __restrict__ AGG, int e) {
    constexpr int QUADS = C >> 2;
    long long idx = (long long)blockIdx.x * blockDim.x + threadIdx.x;
    long long total = (long long)e * QUADS;
    if (idx >= total) return;
    int ei = (int)(idx / QUADS);
    int q  = (int)(idx % QUADS);
    int s = src[ei], d = dst[ei];
    float nrm = dinv[s] * dinv[d];
    float4 h = *(const float4*)(H + (size_t)s * C + q * 4);
    float* out = AGG + (size_t)d * C + q * 4;
    atomicAdd(out + 0, h.x * nrm);
    atomicAdd(out + 1, h.y * nrm);
    atomicAdd(out + 2, h.z * nrm);
    atomicAdd(out + 3, h.w * nrm);
}

// ---------------- bias + relu + bf16 convert ----------------
template <int C>
__global__ void bias_relu_bf16_kernel(const float* __restrict__ AGG, const float* __restrict__ b,
                                      unsigned short* __restrict__ out, int n_nodes) {
    long long idx = (long long)blockIdx.x * blockDim.x + threadIdx.x;
    long long total = (long long)n_nodes * C;
    if (idx >= total) return;
    int c = (int)(idx % C);
    float v = AGG[idx] + b[c];
    v = v > 0.0f ? v : 0.0f;
    out[idx] = f32_to_bf16_bits(v);
}

// ---------------- pooling ----------------
__global__ void zero_f32_kernel(float* __restrict__ p, int n) {
    int i = blockIdx.x * blockDim.x + threadIdx.x;
    if (i < n) p[i] = 0.0f;
}

__global__ __launch_bounds__(OUT_C)
void pool_partial_kernel(const float* __restrict__ H, float* __restrict__ pooled,
                         int rows_per_block, int n_nodes) {
    int c = threadIdx.x;
    int r0 = blockIdx.x * rows_per_block;
    int r1 = r0 + rows_per_block; if (r1 > n_nodes) r1 = n_nodes;
    float s = 0.0f;
    for (int r = r0; r < r1; ++r) s += H[(size_t)r * OUT_C + c];
    atomicAdd(&pooled[c], s);
}

__global__ __launch_bounds__(OUT_C)
void final_fc_kernel(const float* __restrict__ pooled, const float* __restrict__ b2,
                     const float* __restrict__ fcw, const float* __restrict__ fcb,
                     float* __restrict__ out, float inv_nodes) {
    __shared__ float red[OUT_C];
    int c = threadIdx.x;
    // mean(AGG2 + b2) = mean(AGG2) + b2 (bias fused here)
    red[c] = (pooled[c] * inv_nodes + b2[c]) * fcw[c];
    __syncthreads();
    for (int s = OUT_C / 2; s > 0; s >>= 1) {
        if (c < s) red[c] += red[c + s];
        __syncthreads();
    }
    if (c == 0) out[0] = red[0] + fcb[0];
}

// ---------------- host launcher ----------------
static inline size_t align256(size_t x) { return (x + 255) & ~(size_t)255; }

extern "C" void kernel_launch(void* const* d_in, const int* in_sizes, int n_in,
                              void* d_out, int out_size, void* d_ws, size_t ws_size,
                              hipStream_t stream) {
    const float* x    = (const float*)d_in[0];
    const int*   ei   = (const int*)d_in[1];
    const float* W1   = (const float*)d_in[2];
    const float* b1   = (const float*)d_in[3];
    const float* W2   = (const float*)d_in[4];
    const float* b2   = (const float*)d_in[5];
    const float* fcw  = (const float*)d_in[6];
    const float* fcb  = (const float*)d_in[7];
    float*       out  = (float*)d_out;

    const int NN = in_sizes[0] / IN_C;        // 10000
    const int E  = in_sizes[1] / 2;           // 640000
    const int* src = ei;
    const int* dst = ei + E;

    // workspace carve-out
    char* p = (char*)d_ws;
    size_t off = 0;
    auto take = [&](size_t bytes) { char* q = p + off; off += align256(bytes); return q; };
    float*          deg    = (float*)take(sizeof(float) * NN);
    float*          dinv   = (float*)take(sizeof(float) * NN);
    unsigned short* Xbf    = (unsigned short*)take(sizeof(unsigned short) * (size_t)NN * IN_C);
    unsigned short* W1bfT  = (unsigned short*)take(sizeof(unsigned short) * IN_C * HID_C);
    unsigned short* W2bfT  = (unsigned short*)take(sizeof(unsigned short) * HID_C * OUT_C);
    float*          H1     = (float*)take(sizeof(float) * (size_t)NN * HID_C);
    float*          AGG1   = (float*)take(sizeof(float) * (size_t)NN * HID_C);
    unsigned short* A1bf   = (unsigned short*)take(sizeof(unsigned short) * (size_t)NN * HID_C);
    float*          H2     = (float*)take(sizeof(float) * (size_t)NN * OUT_C);
    float*          AGG2   = (float*)take(sizeof(float) * (size_t)NN * OUT_C);
    float*          pooled = (float*)take(sizeof(float) * OUT_C);
    (void)ws_size;

    const int T = 256;
    auto blocks = [&](long long n) { return (unsigned)((n + T - 1) / T); };

    // 1) normalization coefficients
    deg_init_kernel<<<blocks(NN), T, 0, stream>>>(deg, NN);
    deg_count_kernel<<<blocks(E), T, 0, stream>>>(dst, deg, E);
    dinv_kernel<<<blocks(NN), T, 0, stream>>>(deg, dinv, NN);

    // 2) bf16 operands (weights transposed -> K-contiguous for the GEMM)
    to_bf16_kernel<<<blocks((long long)NN * IN_C), T, 0, stream>>>(x, Xbf, NN * IN_C);
    wT_to_bf16_kernel<IN_C, HID_C><<<blocks(IN_C * HID_C), T, 0, stream>>>(W1, W1bfT);
    wT_to_bf16_kernel<HID_C, OUT_C><<<blocks(HID_C * OUT_C), T, 0, stream>>>(W2, W2bfT);

    // 3) layer 1: H1 = X @ W1  (bf16 WMMA, f32 accumulate; TDM-staged LDS strips)
    {
        dim3 g(HID_C / 64, (unsigned)(NN / 16));
        wmma_gemm_kernel<IN_C, HID_C><<<g, 128, 0, stream>>>(Xbf, W1bfT, H1, NN);
    }
    self_loop_init_kernel<HID_C><<<blocks((long long)NN * HID_C), T, 0, stream>>>(H1, dinv, AGG1, NN);
    agg_edges_kernel<HID_C><<<blocks((long long)E * (HID_C / 4)), T, 0, stream>>>(H1, src, dst, dinv, AGG1, E);
    bias_relu_bf16_kernel<HID_C><<<blocks((long long)NN * HID_C), T, 0, stream>>>(AGG1, b1, A1bf, NN);

    // 4) layer 2: H2 = relu(agg1) @ W2
    {
        dim3 g(OUT_C / 64, (unsigned)(NN / 16));
        wmma_gemm_kernel<HID_C, OUT_C><<<g, 128, 0, stream>>>(A1bf, W2bfT, H2, NN);
    }
    self_loop_init_kernel<OUT_C><<<blocks((long long)NN * OUT_C), T, 0, stream>>>(H2, dinv, AGG2, NN);
    agg_edges_kernel<OUT_C><<<blocks((long long)E * (OUT_C / 4)), T, 0, stream>>>(H2, src, dst, dinv, AGG2, E);

    // 5) mean-pool + FC (bias b2 fused into final reduction)
    zero_f32_kernel<<<1, OUT_C, 0, stream>>>(pooled, OUT_C);
    {
        const int rows_per_block = 100;
        unsigned nb = (unsigned)((NN + rows_per_block - 1) / rows_per_block);
        pool_partial_kernel<<<nb, OUT_C, 0, stream>>>(AGG2, pooled, rows_per_block, NN);
    }
    final_fc_kernel<<<1, OUT_C, 0, stream>>>(pooled, b2, fcw, fcb, out, 1.0f / (float)NN);
    (void)out_size; (void)n_in;
}
